// SelfAttention_Module_MS_14499809591918
// MI455X (gfx1250) — matrix operand order repaired and checked by reading, not compiled, not run on hardware
//
#include <hip/hip_runtime.h>

// ---------------------------------------------------------------------------
// bs=8, cin=2048, wh=1024, NCLS=21 (pad 32), NIT=4, dilations 6/12/18/24.
// Algebraic reduction: project conv weights with conv1_w first ->
//   Weff[it][tap][c][ic] = sum_d c1w[c,d]*W[d,ic,tap]   (21-ch dilated conv)
//   t1 = conv21(x) + c1w@b2 ;  wei_cum = softmax_c(mean_it(t1@Bn + c1b))
//   pred = sum_n wei[c,n]*t1[it,c,n] + c1b ;  Bn[n,m] = thr(dW)[m,n]/rs[m]
// Layouts: xT[b][p][ic] (f16), BnT[b][m][n] = thr(dW)/rs — every GEMM staging
// thread loads 64B contiguous. GEMMs use a double-buffered LDS pipeline with
// gfx1250 async global->LDS loads (ASYNCcnt) when the builtins are available.
// ---------------------------------------------------------------------------

typedef __attribute__((ext_vector_type(16))) _Float16 v16h;
typedef __attribute__((ext_vector_type(8)))  _Float16 v8h;
typedef __attribute__((ext_vector_type(4)))  _Float16 v4h;
typedef __attribute__((ext_vector_type(8)))  float    v8f;

static __device__ __forceinline__ v16h cat8(v8h lo, v8h hi) {
  return __builtin_shufflevector(lo, hi, 0,1,2,3,4,5,6,7,8,9,10,11,12,13,14,15);
}

// ---- async global->LDS staging (gfx1250 path, ASYNCcnt-tracked) -----------
#if __has_builtin(__builtin_amdgcn_global_load_async_to_lds_b128) && \
    __has_builtin(__builtin_amdgcn_global_load_async_to_lds_b64) &&  \
    __has_builtin(__builtin_amdgcn_s_wait_asynccnt)
#define USE_ASYNC_LDS 1
typedef __attribute__((ext_vector_type(4))) int i32x4_t;
typedef __attribute__((ext_vector_type(2))) int i32x2_t;
#define AS1G __attribute__((address_space(1)))
#define AS3L __attribute__((address_space(3)))
#define CP16(dst, src)                                                        \
  __builtin_amdgcn_global_load_async_to_lds_b128(                             \
      (AS1G i32x4_t*)(const void*)(src), (AS3L i32x4_t*)(void*)(dst), 0, 0)
#define CP8(dst, src)                                                         \
  __builtin_amdgcn_global_load_async_to_lds_b64(                              \
      (AS1G i32x2_t*)(const void*)(src), (AS3L i32x2_t*)(void*)(dst), 0, 0)
#define STAGE_FENCE() __builtin_amdgcn_s_wait_asynccnt(0)
#else
#define USE_ASYNC_LDS 0
#define CP16(dst, src) (*(v8h*)(dst) = *(const v8h*)(src))
#define CP8(dst, src)  (*(v4h*)(dst) = *(const v4h*)(src))
#define STAGE_FENCE()
#endif

// ---------------- x [b][ic][p] -> xT [b][p][ic] (f16), LDS tile transpose --
__global__ void k_cvt_xT(const float* __restrict__ x, _Float16* __restrict__ xT) {
  __shared__ float t[32][33];
  int b = blockIdx.z, p0 = blockIdx.x * 32, c0 = blockIdx.y * 32;
  int tx = threadIdx.x, ty = threadIdx.y;              // block (32,8)
  for (int r = 0; r < 4; ++r) {
    int i = ty + r * 8;
    t[i][tx] = x[((size_t)(b * 2048 + c0 + i)) * 1024 + p0 + tx];
  }
  __syncthreads();
  for (int r = 0; r < 4; ++r) {
    int j = ty + r * 8;
    xT[((size_t)(b * 1024 + p0 + j)) * 2048 + c0 + tx] = (_Float16)t[tx][j];
  }
}

// ---------------- thresholded rowsums of diffW (one wave / row) ------------
__global__ void k_rowsum(const float* __restrict__ dW, float* __restrict__ rs) {
  int wave = threadIdx.x >> 5, lane = threadIdx.x & 31;
  int row = blockIdx.x * 8 + wave;                     // 8192 rows
  const float* p = dW + (size_t)row * 1024;
  float s = 0.f;
  for (int j = lane; j < 1024; j += 32) { float v = p[j]; s += (v > 0.8f) ? v : 0.f; }
  for (int o = 16; o > 0; o >>= 1) s += __shfl_xor(s, o, 32);
  if (lane == 0) rs[row] = s;
}

// ------ BnT[b][m][n] = thr(dW[b,m,n])/rs[b,m]  (original layout, f16) ------
__global__ void k_build_Bn(const float* __restrict__ dW, const float* __restrict__ rs,
                           _Float16* __restrict__ BnT) {
  int i = blockIdx.x * 256 + threadIdx.x;              // 4 elems each, 2M total
  int row = i >> 8;                                    // (i*4)>>10 : b*1024+m
  float inv = 1.f / rs[row];
  float4 v = ((const float4*)dW)[i];
  float a0 = (v.x > 0.8f) ? v.x : 0.f, a1 = (v.y > 0.8f) ? v.y : 0.f;
  float a2 = (v.z > 0.8f) ? v.z : 0.f, a3 = (v.w > 0.8f) ? v.w : 0.f;
  v4h o = { (_Float16)(a0 * inv), (_Float16)(a1 * inv),
            (_Float16)(a2 * inv), (_Float16)(a3 * inv) };
  ((v4h*)BnT)[i] = o;
}

// -------- Weff[it][tap][c(32)][ic] = sum_d c1w[it][c][d] * W2[it][d][ic][tap]
__global__ __launch_bounds__(256) void k_weff(const float* __restrict__ W2,
                                              const float* __restrict__ C1W,
                                              _Float16* __restrict__ Weff) {
  const int it = blockIdx.y;
  const int j  = blockIdx.x * 256 + threadIdx.x;       // 0..18431 = ic*9+tap
  __shared__ float cw[21 * 256];
  float acc[21];
#pragma unroll
  for (int c = 0; c < 21; ++c) acc[c] = 0.f;
  const float* wbase = W2 + (size_t)it * 1024 * 18432 + j;
  const float* cbase = C1W + (size_t)it * 21 * 1024;
  for (int d0 = 0; d0 < 1024; d0 += 256) {
    __syncthreads();
    for (int e = threadIdx.x; e < 21 * 256; e += 256) {
      int c = e >> 8, dd = e & 255;
      cw[e] = cbase[c * 1024 + d0 + dd];
    }
    __syncthreads();
    for (int dd = 0; dd < 256; ++dd) {
      float wv = wbase[(size_t)(d0 + dd) * 18432];
      if ((dd & 63) == 0) __builtin_prefetch(&wbase[(size_t)(d0 + dd + 64) * 18432], 0, 1);
#pragma unroll
      for (int c = 0; c < 21; ++c) acc[c] += cw[c * 256 + dd] * wv;
    }
  }
  int ic = j / 9, tap = j - ic * 9;
  _Float16* out = Weff + ((size_t)(it * 9 + tap) * 32) * 2048 + ic;
#pragma unroll
  for (int c = 0; c < 32; ++c) out[(size_t)c * 2048] = (_Float16)(c < 21 ? acc[c] : 0.f);
}

// -------- beff[it][c(32)] = sum_d c1w[it][c][d] * conv2d_b[it][d] ----------
__global__ void k_beff(const float* __restrict__ C1W, const float* __restrict__ B2,
                       float* __restrict__ beff) {
  int it = threadIdx.x >> 5, c = threadIdx.x & 31;     // 128 threads
  float a = 0.f;
  if (c < 21) {
    const float* cw = C1W + (size_t)(it * 21 + c) * 1024;
    const float* bb = B2 + it * 1024;
    for (int d = 0; d < 1024; ++d) a += cw[d] * bb[d];
  }
  beff[it * 32 + c] = a;
}

// ---------------------------------------------------------------------------
// WMMA GEMM core: 256 thr (8 waves), tile M=32 x N=256, K=32 per chunk,
// double-buffered LDS (46 KB total). Rows padded to 40 halves (80B) so all
// v8h fragment loads stay 16B aligned. B held transposed in LDS: Bsh[n][k].
// Pipeline: stage(ck+1 -> buf^1) async, compute(buf), wait-async, barrier.
// ---------------------------------------------------------------------------
#define WMMA_F16(a, b, c) \
  __builtin_amdgcn_wmma_f32_16x16x32_f16(false, (a), false, (b), (short)0, (c), false, false)

#define GEMM_FRAGS_AND_MMA(A_, B_, ACC)                                                 \
  do {                                                                                  \
    const int r0 = lane & 15, kh = lane >> 4;                                           \
    v16h a0 = cat8(*(const v8h*)&A_[r0][kh * 8],                                        \
                   *(const v8h*)&A_[r0][16 + kh * 8]);                                  \
    v16h a1 = cat8(*(const v8h*)&A_[16 + r0][kh * 8],                                   \
                   *(const v8h*)&A_[16 + r0][16 + kh * 8]);                             \
    v16h b0 = cat8(*(const v8h*)&B_[nb + r0][kh * 8],                                   \
                   *(const v8h*)&B_[nb + r0][16 + kh * 8]);                             \
    v16h b1 = cat8(*(const v8h*)&B_[nb + 16 + r0][kh * 8],                              \
                   *(const v8h*)&B_[nb + 16 + r0][16 + kh * 8]);                        \
    ACC[0][0] = WMMA_F16(a0, b0, ACC[0][0]);                                            \
    ACC[0][1] = WMMA_F16(a0, b1, ACC[0][1]);                                            \
    ACC[1][0] = WMMA_F16(a1, b0, ACC[1][0]);                                            \
    ACC[1][1] = WMMA_F16(a1, b1, ACC[1][1]);                                            \
  } while (0)

// -------- main kernel: 21-ch dilated conv as 9-tap GEMM, via WMMA ----------
__global__ __launch_bounds__(256) void k_conv_gemm(
    const _Float16* __restrict__ xT, const _Float16* __restrict__ Weff,
    const float* __restrict__ beff, float* __restrict__ t1f, _Float16* __restrict__ t1h) {
  __shared__ __align__(16) _Float16 Ash[2][32][40];
  __shared__ __align__(16) _Float16 Bsh[2][256][40];
  const int it = blockIdx.y >> 3, b = blockIdx.y & 7;
  const int n0 = blockIdx.x * 256;
  const int tid = threadIdx.x, lane = tid & 31, wave = tid >> 5;
  const int nb = wave * 32;
  const int dil = 6 * (it + 1);

  v8f acc[2][2];
#pragma unroll
  for (int i = 0; i < 2; ++i)
#pragma unroll
    for (int j = 0; j < 2; ++j) acc[i][j] = (v8f){};

  const int p = n0 + tid, py = p >> 5, px = p & 31;    // this thread's B column
  const _Float16* xb  = xT + (size_t)b * 1024 * 2048;
  const _Float16* wt0 = Weff + (size_t)it * 9 * 32 * 2048;

  // chunk ck in [0,576): tap = ck>>6, kc = (ck&63)*32
  auto stage = [&](int ck, int buf) {
    const int tap = ck >> 6, kc = (ck & 63) << 5;
    {  // A: 32x32 halves, 8B per thread
      const int e = tid * 4, m = e >> 5, k = e & 31;
      CP8(&Ash[buf][m][k], wt0 + (size_t)tap * 32 * 2048 + (size_t)m * 2048 + kc + k);
    }
    const int dy = tap / 3 - 1, dx = tap % 3 - 1;
    const int sy = py + dil * dy, sx = px + dil * dx;
    const bool valid = ((unsigned)sy < 32u) && ((unsigned)sx < 32u);
    if (valid) {  // B: 64B contiguous per thread into its LDS row
      const _Float16* bsrc = xb + (size_t)((sy << 5) + sx) * 2048 + kc;
      _Float16* bdst = &Bsh[buf][tid][0];
#pragma unroll
      for (int q = 0; q < 4; ++q) CP16(bdst + q * 8, bsrc + q * 8);
    } else if ((ck & 63) < 2) {  // zero this row in BOTH buffers once per tap
      v8h z = {};
      v8h* dst = (v8h*)&Bsh[buf][tid][0];
#pragma unroll
      for (int q = 0; q < 4; ++q) dst[q] = z;
    }
  };

  const int NCK = 9 * 64;
  stage(0, 0);
  STAGE_FENCE();
  __syncthreads();
  for (int ck = 0; ck < NCK; ++ck) {
    const int buf = ck & 1;
    if (ck + 1 < NCK) stage(ck + 1, buf ^ 1);
    GEMM_FRAGS_AND_MMA(Ash[buf], Bsh[buf], acc);
    STAGE_FENCE();
    __syncthreads();
  }

  // epilogue: C layout — lane<16: M=r, N=lane; lane>=16: M=r+8, N=lane-16
  const int mh = lane >> 4, col = lane & 15;
#pragma unroll
  for (int i = 0; i < 2; ++i)
#pragma unroll
    for (int j = 0; j < 2; ++j)
#pragma unroll
      for (int r = 0; r < 8; ++r) {
        int c = i * 16 + mh * 8 + r;
        int nn = n0 + nb + j * 16 + col;
        float v = acc[i][j][r] + beff[it * 32 + c];
        t1h[((size_t)(it * 8 + b) * 32 + c) * 1024 + nn] = (_Float16)v;
        if (c < 21) t1f[((size_t)(it * 8 + b) * 21 + c) * 1024 + nn] = v;
      }
}

// -------- wei_cum = mean_it( t1[it] @ Bn + c1b[it] ), via WMMA -------------
__global__ __launch_bounds__(256) void k_o2_gemm(
    const _Float16* __restrict__ t1h, const _Float16* __restrict__ BnT,
    const float* __restrict__ C1B, float* __restrict__ wcum) {
  __shared__ __align__(16) _Float16 Ash[2][32][40];
  __shared__ __align__(16) _Float16 Bsh[2][256][40];
  const int b = blockIdx.y, m0 = blockIdx.x * 256;
  const int tid = threadIdx.x, lane = tid & 31, wave = tid >> 5;
  const int nb = wave * 32;

  v8f acc[2][2];
#pragma unroll
  for (int i = 0; i < 2; ++i)
#pragma unroll
    for (int j = 0; j < 2; ++j) acc[i][j] = (v8f){};

  const _Float16* brow = BnT + ((size_t)(b * 1024 + m0 + tid)) * 1024; // this col's row

  // chunk ck in [0,128): it = ck>>5, kc = (ck&31)*32  (K = NIT*1024 folded)
  auto stage = [&](int ck, int buf) {
    const int it = ck >> 5, kc = (ck & 31) << 5;
    {  // A
      const int e = tid * 4, m = e >> 5, k = e & 31;
      CP8(&Ash[buf][m][k], t1h + ((size_t)(it * 8 + b) * 32 + m) * 1024 + kc + k);
    }
    {  // B: 64B contiguous per thread
      const _Float16* bsrc = brow + kc;
      _Float16* bdst = &Bsh[buf][tid][0];
#pragma unroll
      for (int q = 0; q < 4; ++q) CP16(bdst + q * 8, bsrc + q * 8);
    }
  };

  const int NCK = 128;
  stage(0, 0);
  STAGE_FENCE();
  __syncthreads();
  for (int ck = 0; ck < NCK; ++ck) {
    const int buf = ck & 1;
    if (ck + 1 < NCK) stage(ck + 1, buf ^ 1);
    GEMM_FRAGS_AND_MMA(Ash[buf], Bsh[buf], acc);
    STAGE_FENCE();
    __syncthreads();
  }

  const int mh = lane >> 4, col = lane & 15;
#pragma unroll
  for (int i = 0; i < 2; ++i)
#pragma unroll
    for (int j = 0; j < 2; ++j)
#pragma unroll
      for (int r = 0; r < 8; ++r) {
        int c = i * 16 + mh * 8 + r;
        if (c >= 21) continue;
        int m = m0 + nb + j * 16 + col;
        float bias = 0.25f * (C1B[c] + C1B[21 + c] + C1B[42 + c] + C1B[63 + c]);
        wcum[((size_t)(b * 21 + c)) * 1024 + m] = 0.25f * acc[i][j][r] + bias;
      }
}

// -------- softmax over classes (x2): wei_cum output + inp = softmax^2 ------
__global__ void k_softmax_cls(const float* __restrict__ wcum,
                              float* __restrict__ out_wc, float* __restrict__ inp) {
  int g = blockIdx.x * 256 + threadIdx.x;   // 8192 (b,m) positions
  int b = g >> 10, m = g & 1023;
  float v[21], mx = -1e30f, s;
#pragma unroll
  for (int c = 0; c < 21; ++c) { v[c] = wcum[((size_t)(b * 21 + c)) * 1024 + m]; mx = fmaxf(mx, v[c]); }
  s = 0.f;
#pragma unroll
  for (int c = 0; c < 21; ++c) { v[c] = __expf(v[c] - mx); s += v[c]; }
  float inv = 1.f / s; mx = -1e30f;
#pragma unroll
  for (int c = 0; c < 21; ++c) { v[c] *= inv; out_wc[((size_t)(b * 21 + c)) * 1024 + m] = v[c]; mx = fmaxf(mx, v[c]); }
  s = 0.f;
#pragma unroll
  for (int c = 0; c < 21; ++c) { v[c] = __expf(v[c] - mx); s += v[c]; }
  inv = 1.f / s;
#pragma unroll
  for (int c = 0; c < 21; ++c) inp[((size_t)(b * 21 + c)) * 1024 + m] = v[c] * inv;
}

// -------- spatial softmax pooling + final prediction -----------------------
static __device__ __forceinline__ float blockReduce(float v, float* s, bool ismax) {
  for (int o = 16; o > 0; o >>= 1) {
    float t = __shfl_xor(v, o, 32);
    v = ismax ? fmaxf(v, t) : v + t;
  }
  int lane = threadIdx.x & 31, w = threadIdx.x >> 5;
  __syncthreads();
  if (lane == 0) s[w] = v;
  __syncthreads();
  float r = s[0];
  for (int i = 1; i < 8; ++i) r = ismax ? fmaxf(r, s[i]) : r + s[i];
  return r;
}

__global__ __launch_bounds__(256) void k_wei(
    const float* __restrict__ inp, const float* __restrict__ t1f,
    const float* __restrict__ PW, const float* __restrict__ C1B,
    float* __restrict__ out_prob, float* __restrict__ out_wei) {
  __shared__ float red[8];
  const int bc = blockIdx.x;                  // = b*21 + c, 168 blocks
  const int b = bc / 21, c = bc - b * 21;
  const int tid = threadIdx.x;
  float iv[4], wlast[4] = {0.f, 0.f, 0.f, 0.f};
  const float* ip = inp + (size_t)bc * 1024;
#pragma unroll
  for (int k = 0; k < 4; ++k) iv[k] = ip[tid + k * 256];
  float prob_acc = 0.f;
  for (int it = 0; it < 4; ++it) {
    float e = __expf(PW[it * 21 + c]);
    float sv[4], mx = -1e30f;
#pragma unroll
    for (int k = 0; k < 4; ++k) { sv[k] = iv[k] * e; mx = fmaxf(mx, sv[k]); }
    float M = blockReduce(mx, red, true);
    float ls = 0.f;
#pragma unroll
    for (int k = 0; k < 4; ++k) { sv[k] = __expf(sv[k] - M); ls += sv[k]; }
    float S = blockReduce(ls, red, false);
    float inv = 1.f / S, pd = 0.f;
    const float* t1row = t1f + ((size_t)(it * 8 + b) * 21 + c) * 1024;
#pragma unroll
    for (int k = 0; k < 4; ++k) {
      float w = sv[k] * inv;
      pd += w * t1row[tid + k * 256];
      if (it == 3) wlast[k] = w;
    }
    pd = blockReduce(pd, red, false);
    prob_acc += 0.25f * (pd + C1B[it * 21 + c]);
  }
#pragma unroll
  for (int k = 0; k < 4; ++k) out_wei[(size_t)bc * 1024 + tid + k * 256] = wlast[k];
  if (tid == 0) out_prob[bc] = 1.f / (1.f + __expf(-prob_acc));
}

// ---------------------------------------------------------------------------
extern "C" void kernel_launch(void* const* d_in, const int* in_sizes, int n_in,
                              void* d_out, int out_size, void* d_ws, size_t ws_size,
                              hipStream_t stream) {
  const float* x   = (const float*)d_in[0];   // [8,2048,32,32]
  const float* dW  = (const float*)d_in[1];   // [8,1024,1024]
  const float* W2  = (const float*)d_in[2];   // [4,1024,2048,3,3]
  const float* B2  = (const float*)d_in[3];   // [4,1024]
  const float* C1W = (const float*)d_in[4];   // [4,21,1024,1,1]
  const float* C1B = (const float*)d_in[5];   // [4,21]
  const float* PW  = (const float*)d_in[6];   // [4,1,21]
  float* out = (float*)d_out;                 // prob(168) | wei_cum(172032) | wei(172032)
  (void)in_sizes; (void)n_in; (void)out_size; (void)ws_size;

  char* ws = (char*)d_ws;
  size_t off = 0;
  auto alloc = [&](size_t bytes) -> void* {
    void* p = ws + off;
    off = (off + bytes + 255) & ~(size_t)255;
    return p;
  };
  float*    rs   = (float*)   alloc((size_t)8192 * 4);
  _Float16* BnT  = (_Float16*)alloc((size_t)8 * 1024 * 1024 * 2);    // 16 MB
  _Float16* xT   = (_Float16*)alloc((size_t)8 * 1024 * 2048 * 2);    // 32 MB
  _Float16* Weff = (_Float16*)alloc((size_t)4 * 9 * 32 * 2048 * 2);  // 4.5 MB
  float*    beff = (float*)   alloc((size_t)4 * 32 * 4);
  float*    t1f  = (float*)   alloc((size_t)4 * 8 * 21 * 1024 * 4);  // 2.75 MB
  _Float16* t1h  = (_Float16*)alloc((size_t)4 * 8 * 32 * 1024 * 2);  // 2.1 MB
  float*    wcum = (float*)   alloc((size_t)8 * 21 * 1024 * 4);
  float*    inp  = (float*)   alloc((size_t)8 * 21 * 1024 * 4);

  k_cvt_xT<<<dim3(32, 64, 8), dim3(32, 8), 0, stream>>>(x, xT);
  k_rowsum<<<1024, 256, 0, stream>>>(dW, rs);
  k_build_Bn<<<8192, 256, 0, stream>>>(dW, rs, BnT);
  k_weff<<<dim3(72, 4), 256, 0, stream>>>(W2, C1W, Weff);
  k_beff<<<1, 128, 0, stream>>>(C1W, B2, beff);
  k_conv_gemm<<<dim3(4, 32), 256, 0, stream>>>(xT, Weff, beff, t1f, t1h);
  k_o2_gemm<<<dim3(4, 8), 256, 0, stream>>>(t1h, BnT, C1B, wcum);
  k_softmax_cls<<<32, 256, 0, stream>>>(wcum, out + 168, inp);
  k_wei<<<168, 256, 0, stream>>>(inp, t1f, PW, C1B, out, out + 168 + 172032);
}
